// TvpAttention_2645699854987
// MI455X (gfx1250) — matrix operand order repaired and checked by reading, not compiled, not run on hardware
//
#include <hip/hip_runtime.h>

// ---------------- problem constants ----------------
#define BATCH 4
#define SEQ   2048
#define HDIM  768
#define NHEAD 12
#define HEADD 64
#define MROWS (BATCH * SEQ)   // 8192

typedef __attribute__((ext_vector_type(8)))  float  v8f;
typedef __attribute__((ext_vector_type(16))) __bf16 v16bf;

// ---------------- Tensor Data Mover (TDM) 1-D tile copy ----------------
#if __has_builtin(__builtin_amdgcn_tensor_load_to_lds) && \
    __has_builtin(__builtin_amdgcn_s_wait_tensorcnt)
#define USE_TDM 1
#endif

typedef __attribute__((ext_vector_type(4))) unsigned int u32x4;
typedef __attribute__((ext_vector_type(8))) int          i32x8;
typedef __attribute__((ext_vector_type(4))) int          i32x4;

#ifdef USE_TDM
// Copy `nelems` contiguous bf16 elements from global to LDS via the TDM.
// D# layout per CDNA5 ISA:
//   g0: [1:0]=count=1 | lds_addr | global_addr lo | global_addr[56:32]+type=2
//   g1: data_size=1 (2B); tensor_dim0 = tile_dim0 = nelems; tensor_dim1 = 1
__device__ __forceinline__ void tdm_copy_bf16(const void* gsrc,
                                              void* lds_dst,
                                              unsigned int nelems) {
    unsigned long long ga = (unsigned long long)(size_t)gsrc;
    u32x4 g0;
    g0[0] = 1u;                                         // count=1, user D#
    g0[1] = (unsigned int)(size_t)lds_dst;              // lds_addr (bytes)
    g0[2] = (unsigned int)ga;                           // global_addr[31:0]
    g0[3] = (unsigned int)((ga >> 32) & 0x1FFFFFFu)     // global_addr[56:32]
          | (2u << 30);                                 // type = 2 ("image")
    i32x8 g1;
    g1[0] = 0x00010000;                                 // data_size=1 (2 bytes)
    g1[1] = (int)((nelems & 0xFFFFu) << 16);            // tensor_dim0[15:0]
    g1[2] = (int)((nelems >> 16) & 0xFFFFu) | (1 << 16);// dim0 hi | tensor_dim1=1
    g1[3] = (int)((nelems & 0xFFFFu) << 16);            // tile_dim0
    g1[4] = 0;                                          // tile_dim1/2 unused
    g1[5] = (int)nelems;                                // tensor_dim0_stride lo
    g1[6] = 0;
    g1[7] = 0;
    i32x4 z4 = {0, 0, 0, 0};
#if __clang_major__ >= 23
    i32x8 z8 = {0, 0, 0, 0, 0, 0, 0, 0};
    __builtin_amdgcn_tensor_load_to_lds(g0, g1, z4, z4, z8, 0);
#else
    __builtin_amdgcn_tensor_load_to_lds(g0, g1, z4, z4, 0);
#endif
}
#endif

// =====================================================================
// Kernel 1: out[m,n] = (sum_k X[m,k]*W[n,k] + bias[n]) * out_scale
// Block tile 128(M) x 64(N), K-step 32. 8 waves; each wave owns a
// 16x64 strip: one A-fragment reused across 4 B-fragments = 4 WMMA.
// Out bf16 in [B, NH, S, HD] layout. fp32->bf16 via hardware convert.
// =====================================================================
__global__ __launch_bounds__(256) void qkv_gemm(
    const float* __restrict__ X, const float* __restrict__ W,
    const float* __restrict__ bias, __bf16* __restrict__ Out,
    float out_scale)
{
    __shared__ __bf16 As[128][32];  // 8 KB
    __shared__ __bf16 Bs[64][32];   // 4 KB

    const int tid  = threadIdx.x;
    const int lane = tid & 31;
    const int wave = tid >> 5;
    const int lh   = lane >> 4;
    const int l16  = lane & 15;
    const int m0   = blockIdx.x * 128;
    const int n0   = blockIdx.y * 64;

    v8f acc[4] = {};

    for (int k0 = 0; k0 < HDIM; k0 += 32) {
        __syncthreads();
        // stage A: 128x32 fp32 -> bf16 (16 per thread, 64B contiguous)
        #pragma unroll
        for (int i = 0; i < 16; ++i) {
            int idx = tid * 16 + i;
            int row = idx >> 5, col = idx & 31;
            As[row][col] = (__bf16)X[(size_t)(m0 + row) * HDIM + k0 + col];
        }
        // stage B: 64x32 of W (8 per thread)
        #pragma unroll
        for (int i = 0; i < 8; ++i) {
            int idx = tid * 8 + i;
            int row = idx >> 5, col = idx & 31;
            Bs[row][col] = (__bf16)W[(size_t)(n0 + row) * HDIM + k0 + col];
        }
        // prefetch next K panel (global_prefetch_b8)
        if (k0 + 32 < HDIM) {
            __builtin_prefetch(X + (size_t)(m0 + (tid >> 1)) * HDIM + k0 + 32, 0, 1);
            __builtin_prefetch(W + (size_t)(n0 + (tid >> 2)) * HDIM + k0 + 32, 0, 1);
        }
        __syncthreads();

        // one A fragment, four B fragments, then 4 back-to-back WMMAs
        v16bf a;
        #pragma unroll
        for (int e = 0; e < 16; ++e) {
            int ka = lh * 16 + e;
            a[e] = As[wave * 16 + l16][ka];
        }
        v16bf bm[4];
        #pragma unroll
        for (int nw = 0; nw < 4; ++nw)
            #pragma unroll
            for (int e = 0; e < 16; ++e)
                bm[nw][e] = Bs[nw * 16 + l16][lh * 16 + e];
        #pragma unroll
        for (int nw = 0; nw < 4; ++nw)
            acc[nw] = __builtin_amdgcn_wmma_f32_16x16x32_bf16(
                          false, a, false, bm[nw], (short)0, acc[nw], false, false);
    }

    #pragma unroll
    for (int nw = 0; nw < 4; ++nw) {
        const int ncol = n0 + nw * 16 + l16;
        const float bi = bias[ncol];
        const int hh = ncol / HEADD, dd = ncol % HEADD;
        #pragma unroll
        for (int r = 0; r < 8; ++r) {
            int m = m0 + wave * 16 + r + 8 * lh;
            int bb = m / SEQ, ss = m % SEQ;
            float v = (acc[nw][r] + bi) * out_scale;
            Out[((size_t)(bb * NHEAD + hh) * SEQ + ss) * HEADD + dd] = (__bf16)v;
        }
    }
}

// =====================================================================
// Kernel 2: flash-style attention per (b, head).
// Q/K/V bf16 [B,NH,S,HD]. Each wave owns 16 query rows, loops over
// 32-key tiles. K/V tiles staged into LDS by the Tensor Data Mover
// (one issuing wave + s_wait_tensorcnt + workgroup barrier); vector
// load fallback if the TDM builtin is unavailable.
// =====================================================================
__global__ __launch_bounds__(256) void attn(
    const __bf16* __restrict__ Qb,
    const __bf16* __restrict__ Kb,
    const __bf16* __restrict__ Vb,
    const float* __restrict__ mask,            // [B,1,1,S]
    __bf16* __restrict__ CtxOut)               // bf16 [MROWS, HDIM]
{
    __shared__ __bf16 Ks[32][64];      // 4 KB
    __shared__ __bf16 Vs[32][64];      // 4 KB
    __shared__ __bf16 Pw[8][16][32];   // 8 KB (wave-private P tiles)

    const int tid  = threadIdx.x;
    const int lane = tid & 31;
    const int wave = tid >> 5;
    const int lh   = lane >> 4;
    const int l16  = lane & 15;
    const int bidx = blockIdx.z;
    const int head = blockIdx.y;
    const int bh   = bidx * NHEAD + head;
    const int qbase = blockIdx.x * 128 + wave * 16;

    // Q fragments for K-chunks d=0..31 and d=32..63 (A-layout gather)
    v16bf aq0, aq1;
    {
        const __bf16* Qrow = Qb + ((size_t)bh * SEQ + qbase + l16) * HEADD;
        #pragma unroll
        for (int e = 0; e < 16; ++e) {
            int ka = lh * 16 + e;
            aq0[e] = Qrow[ka];
            aq1[e] = Qrow[32 + ka];
        }
    }

    v8f acc[4] = {};                 // ctx accum: 4 chunks of 16 d-cols
    float m_i[8], l_i[8];
    #pragma unroll
    for (int r = 0; r < 8; ++r) { m_i[r] = -INFINITY; l_i[r] = 0.0f; }

    for (int kt = 0; kt < SEQ / 32; ++kt) {
        __syncthreads();
        const __bf16* Ksrc = Kb + ((size_t)bh * SEQ + kt * 32) * HEADD;
        const __bf16* Vsrc = Vb + ((size_t)bh * SEQ + kt * 32) * HEADD;
#ifdef USE_TDM
        if (wave == 0) {
            tdm_copy_bf16(Ksrc, &Ks[0][0], 32 * HEADD);
            tdm_copy_bf16(Vsrc, &Vs[0][0], 32 * HEADD);
            __builtin_amdgcn_s_wait_tensorcnt(0);
        }
#else
        {
            const unsigned int* Ku = reinterpret_cast<const unsigned int*>(Ksrc);
            const unsigned int* Vu = reinterpret_cast<const unsigned int*>(Vsrc);
            unsigned int* Kdst = reinterpret_cast<unsigned int*>(&Ks[0][0]);
            unsigned int* Vdst = reinterpret_cast<unsigned int*>(&Vs[0][0]);
            #pragma unroll
            for (int i = 0; i < 4; ++i) {
                Kdst[tid + 256 * i] = Ku[tid + 256 * i];
                Vdst[tid + 256 * i] = Vu[tid + 256 * i];
            }
        }
#endif
        __syncthreads();

        // ---- scores S = Q K^T : two 16x16 tiles (keys 0-15, 16-31) ----
        v8f s0 = {}, s1 = {};
        {
            v16bf b00, b01, b10, b11;   // B[k][n] = K[n][k]
            #pragma unroll
            for (int e = 0; e < 16; ++e) {
                int kb = lh * 16 + e;
                b00[e] = Ks[l16][kb];
                b01[e] = Ks[l16][32 + kb];
                b10[e] = Ks[16 + l16][kb];
                b11[e] = Ks[16 + l16][32 + kb];
            }
            s0 = __builtin_amdgcn_wmma_f32_16x16x32_bf16(false, aq0, false, b00, (short)0, s0, false, false);
            s0 = __builtin_amdgcn_wmma_f32_16x16x32_bf16(false, aq1, false, b01, (short)0, s0, false, false);
            s1 = __builtin_amdgcn_wmma_f32_16x16x32_bf16(false, aq0, false, b10, (short)0, s1, false, false);
            s1 = __builtin_amdgcn_wmma_f32_16x16x32_bf16(false, aq1, false, b11, (short)0, s1, false, false);
        }

        const float mk0 = mask[(size_t)bidx * SEQ + kt * 32 + l16];
        const float mk1 = mask[(size_t)bidx * SEQ + kt * 32 + 16 + l16];

        // ---- online softmax update (row = r + 8*lh, cols across 16 lanes) ----
        #pragma unroll
        for (int r = 0; r < 8; ++r) {
            float a0 = s0[r] + mk0;
            float a1 = s1[r] + mk1;
            float t = fmaxf(a0, a1);
            #pragma unroll
            for (int mmask = 8; mmask >= 1; mmask >>= 1)
                t = fmaxf(t, __shfl_xor(t, mmask, 16));
            float mnew = fmaxf(m_i[r], t);
            float corr = __expf(m_i[r] - mnew);
            float p0 = __expf(a0 - mnew);
            float p1 = __expf(a1 - mnew);
            float ps = p0 + p1;
            #pragma unroll
            for (int mmask = 8; mmask >= 1; mmask >>= 1)
                ps += __shfl_xor(ps, mmask, 16);
            l_i[r] = l_i[r] * corr + ps;
            m_i[r] = mnew;
            acc[0][r] *= corr; acc[1][r] *= corr;
            acc[2][r] *= corr; acc[3][r] *= corr;
            int prow = r + 8 * lh;
            Pw[wave][prow][l16]      = (__bf16)p0;
            Pw[wave][prow][16 + l16] = (__bf16)p1;
        }

        // ---- reload P as A-fragment (C->A relayout via wave-private LDS) ----
        v16bf ap;
        #pragma unroll
        for (int e = 0; e < 16; ++e) {
            int ka = lh * 16 + e;
            ap[e] = Pw[wave][l16][ka];
        }

        // ---- ctx += P * V  (4 d-chunks of 16): load all B frags, then WMMA ----
        v16bf bv[4];
        #pragma unroll
        for (int dc = 0; dc < 4; ++dc)
            #pragma unroll
            for (int e = 0; e < 16; ++e)
                bv[dc][e] = Vs[lh * 16 + e][dc * 16 + l16];
        #pragma unroll
        for (int dc = 0; dc < 4; ++dc)
            acc[dc] = __builtin_amdgcn_wmma_f32_16x16x32_bf16(
                          false, ap, false, bv[dc], (short)0, acc[dc], false, false);
    }

    // ---- normalize and write ctx ----
    #pragma unroll
    for (int r = 0; r < 8; ++r) {
        float inv = (l_i[r] > 0.0f) ? (1.0f / l_i[r]) : 0.0f;
        int qrow = qbase + r + 8 * lh;
        size_t base = ((size_t)bidx * SEQ + qrow) * HDIM + head * HEADD;
        #pragma unroll
        for (int dc = 0; dc < 4; ++dc)
            CtxOut[base + dc * 16 + l16] = (__bf16)(acc[dc][r] * inv);
    }
}

// =====================================================================
// Kernel 3: H[m,n] = sum_k Ctx[m,k]*Wo[n,k] + bo[n] + resid[m,n] (fp32)
// Same 128x64 tiling, 4 WMMA per wave per K-step. Ctx already bf16.
// =====================================================================
__global__ __launch_bounds__(256) void out_proj(
    const __bf16* __restrict__ Ctx, const float* __restrict__ Wo,
    const float* __restrict__ bo, const float* __restrict__ resid,
    float* __restrict__ Hout)
{
    __shared__ __bf16 As[128][32];
    __shared__ __bf16 Bs[64][32];

    const int tid  = threadIdx.x;
    const int lane = tid & 31;
    const int wave = tid >> 5;
    const int lh   = lane >> 4;
    const int l16  = lane & 15;
    const int m0   = blockIdx.x * 128;
    const int n0   = blockIdx.y * 64;

    v8f acc[4] = {};

    for (int k0 = 0; k0 < HDIM; k0 += 32) {
        __syncthreads();
        {   // A tile: 128x32 bf16 bits as uints (2048 uints, 8 per thread)
            unsigned int* dst = reinterpret_cast<unsigned int*>(&As[0][0]);
            #pragma unroll
            for (int i = 0; i < 8; ++i) {
                int idx = tid * 8 + i;
                int row = idx >> 4, col = idx & 15;
                dst[idx] = reinterpret_cast<const unsigned int*>(
                    Ctx + (size_t)(m0 + row) * HDIM + k0)[col];
            }
        }
        #pragma unroll
        for (int i = 0; i < 8; ++i) {
            int idx = tid * 8 + i;
            int row = idx >> 5, col = idx & 31;
            Bs[row][col] = (__bf16)Wo[(size_t)(n0 + row) * HDIM + k0 + col];
        }
        if (k0 + 32 < HDIM) {
            __builtin_prefetch(Ctx + (size_t)(m0 + (tid >> 1)) * HDIM + k0 + 32, 0, 1);
            __builtin_prefetch(Wo + (size_t)(n0 + (tid >> 2)) * HDIM + k0 + 32, 0, 1);
        }
        __syncthreads();

        v16bf a;
        #pragma unroll
        for (int e = 0; e < 16; ++e) {
            int ka = lh * 16 + e;
            a[e] = As[wave * 16 + l16][ka];
        }
        v16bf bm[4];
        #pragma unroll
        for (int nw = 0; nw < 4; ++nw)
            #pragma unroll
            for (int e = 0; e < 16; ++e)
                bm[nw][e] = Bs[nw * 16 + l16][lh * 16 + e];
        #pragma unroll
        for (int nw = 0; nw < 4; ++nw)
            acc[nw] = __builtin_amdgcn_wmma_f32_16x16x32_bf16(
                          false, a, false, bm[nw], (short)0, acc[nw], false, false);
    }

    #pragma unroll
    for (int nw = 0; nw < 4; ++nw) {
        const int ncol = n0 + nw * 16 + l16;
        const float bi = bo[ncol];
        #pragma unroll
        for (int r = 0; r < 8; ++r) {
            int m = m0 + wave * 16 + r + 8 * lh;
            Hout[(size_t)m * HDIM + ncol] =
                acc[nw][r] + bi + resid[(size_t)m * HDIM + ncol];
        }
    }
}

// =====================================================================
// Kernel 4: row-wise LayerNorm, one block (256 thr) per row of 768.
// =====================================================================
__global__ __launch_bounds__(256) void ln_k(
    const float* __restrict__ X, const float* __restrict__ g,
    const float* __restrict__ be, float* __restrict__ O)
{
    const int row = blockIdx.x;
    const int tid = threadIdx.x;
    const float* x = X + (size_t)row * HDIM;
    float v0 = x[tid], v1 = x[tid + 256], v2 = x[tid + 512];
    float s = v0 + v1 + v2;
    float q = v0 * v0 + v1 * v1 + v2 * v2;
    #pragma unroll
    for (int mmask = 16; mmask >= 1; mmask >>= 1) {
        s += __shfl_xor(s, mmask, 32);
        q += __shfl_xor(q, mmask, 32);
    }
    __shared__ float rs[8], rq[8];
    if ((tid & 31) == 0) { rs[tid >> 5] = s; rq[tid >> 5] = q; }
    __syncthreads();
    s = rs[tid & 7]; q = rq[tid & 7];
    #pragma unroll
    for (int mmask = 4; mmask >= 1; mmask >>= 1) {
        s += __shfl_xor(s, mmask, 8);
        q += __shfl_xor(q, mmask, 8);
    }
    float mean = s * (1.0f / (float)HDIM);
    float var  = q * (1.0f / (float)HDIM) - mean * mean;
    float rstd = rsqrtf(var + 1e-12f);
    float* o = O + (size_t)row * HDIM;
    o[tid]       = (v0 - mean) * rstd * g[tid]       + be[tid];
    o[tid + 256] = (v1 - mean) * rstd * g[tid + 256] + be[tid + 256];
    o[tid + 512] = (v2 - mean) * rstd * g[tid + 512] + be[tid + 512];
}

// =====================================================================
extern "C" void kernel_launch(void* const* d_in, const int* in_sizes, int n_in,
                              void* d_out, int out_size, void* d_ws, size_t ws_size,
                              hipStream_t stream) {
    const float* hidden = (const float*)d_in[0];
    const float* mask   = (const float*)d_in[1];
    const float* Wq = (const float*)d_in[2];
    const float* bq = (const float*)d_in[3];
    const float* Wk = (const float*)d_in[4];
    const float* bk = (const float*)d_in[5];
    const float* Wv = (const float*)d_in[6];
    const float* bv = (const float*)d_in[7];
    const float* Wo = (const float*)d_in[8];
    const float* bo = (const float*)d_in[9];
    const float* ln_g = (const float*)d_in[10];
    const float* ln_b = (const float*)d_in[11];
    float* out = (float*)d_out;

    const size_t nElem = (size_t)MROWS * HDIM;           // 6,291,456
    char* ws = (char*)d_ws;
    __bf16* qbuf = (__bf16*)(ws);
    __bf16* kbuf = qbuf + nElem;
    __bf16* vbuf = kbuf + nElem;
    __bf16* cbuf = vbuf + nElem;
    float*  hbuf = (float*)(cbuf + nElem);

    dim3 gGemm(MROWS / 128, HDIM / 64);    // 64 x 12
    qkv_gemm<<<gGemm, 256, 0, stream>>>(hidden, Wq, bq, qbuf, 0.125f); // 1/sqrt(64)
    qkv_gemm<<<gGemm, 256, 0, stream>>>(hidden, Wk, bk, kbuf, 1.0f);
    qkv_gemm<<<gGemm, 256, 0, stream>>>(hidden, Wv, bv, vbuf, 1.0f);

    dim3 gAttn(SEQ / 128, NHEAD, BATCH);   // 16 x 12 x 4
    attn<<<gAttn, 256, 0, stream>>>(qbuf, kbuf, vbuf, mask, cbuf);

    out_proj<<<gGemm, 256, 0, stream>>>(cbuf, Wo, bo, hidden, hbuf);

    ln_k<<<MROWS, 256, 0, stream>>>(hbuf, ln_g, ln_b, out);
}
